// MoELayer_3521873183473
// MI455X (gfx1250) — matrix operand order repaired and checked by reading, not compiled, not run on hardware
//
#include <hip/hip_runtime.h>
#include <hip/hip_bf16.h>

// ---------------------------------------------------------------------------
// Switch-MoE layer for MI455X (gfx1250): bf16 WMMA + Tensor Data Mover path.
// B=8, N=2048, D=1024, E=8, DFF=4096, CAP=320
// ---------------------------------------------------------------------------

#define MOE_B   8
#define MOE_N   2048
#define MOE_D   1024
#define MOE_E   8
#define MOE_DFF 4096
#define MOE_CAP 320

typedef __attribute__((ext_vector_type(16))) __bf16       v16bf;
typedef __attribute__((ext_vector_type(8)))  float        v8f;
typedef __attribute__((ext_vector_type(4)))  unsigned int u32x4;
typedef __attribute__((ext_vector_type(8)))  int          i32x8;
typedef __attribute__((ext_vector_type(4)))  int          i32x4;

#if __has_builtin(__builtin_amdgcn_tensor_load_to_lds) && \
    __has_builtin(__builtin_amdgcn_s_wait_tensorcnt)
#define MOE_HAVE_TDM 1
#else
#define MOE_HAVE_TDM 0
#endif

__device__ __forceinline__ __bf16 f2bf(float f) {
    // round-to-nearest-even fp32 -> bf16
    unsigned u = __builtin_bit_cast(unsigned, f);
    unsigned r = (u + 0x7FFFu + ((u >> 16) & 1u)) >> 16;
    unsigned short h = (unsigned short)r;
    return __builtin_bit_cast(__bf16, h);
}

#if MOE_HAVE_TDM
// ---------------------------------------------------------------------------
// TDM 2D tile load: tile_k x tile_n (dim0 = K contiguous, elem = 2B bf16),
// row stride = row_stride_elems. LDS layout padded +16B every 64B row
// (pad_interval=16 DWORDs -> code 3, pad_amount=4 DWORDs -> code 3), i.e.
// LDS rows of 40 bf16 elements.  D# packing per ISA 08_async_tensor §8.3/8.4.
// This toolchain exposes the 6-arg builtin:
//   (u32x4 g0, i32x8 g1, i32x4 g2, i32x4 g3, i32x8 pad, i32 cpol)
// ---------------------------------------------------------------------------
__device__ __forceinline__ void tdm_load_tile(unsigned lds_off, const void* gptr,
                                              unsigned tile_k, unsigned tile_n,
                                              unsigned long long row_stride_elems) {
    unsigned long long ga = (unsigned long long)(size_t)gptr;   // 57-bit address
    u32x4 g0;
    g0[0] = 1u;                                        // count=1, user mode
    g0[1] = lds_off;                                   // lds_addr
    g0[2] = (unsigned)(ga & 0xFFFFFFFFu);              // global_addr[31:0]
    g0[3] = (unsigned)((ga >> 32) & 0x01FFFFFFu)       // global_addr[56:32]
            | (2u << 30);                              // type = 2 ("image")
    const unsigned td0 = 0x40000000u;                  // huge tensor dims (no OOB)
    const unsigned td1 = 0x40000000u;
    i32x8 g1;
    g1[0] = (int)((1u << 16)      // data_size = 1 -> 2 bytes
                | (1u << 20)      // pad_enable
                | (3u << 22)      // pad_interval: 2^(3+1)=16 DWORDs (64B)
                | (3u << 25));    // pad_amount: 3+1 = 4 DWORDs (16B)
    g1[1] = (int)((td0 & 0xFFFFu) << 16);                                   // tensor_dim0 lo16
    g1[2] = (int)(((td0 >> 16) & 0xFFFFu) | ((td1 & 0xFFFFu) << 16));       // td0 hi16 | td1 lo16
    g1[3] = (int)(((td1 >> 16) & 0xFFFFu) | ((tile_k & 0xFFFFu) << 16));    // td1 hi16 | tile_dim0
    g1[4] = (int)(tile_n & 0xFFFFu);                                        // tile_dim1 (tile_dim2=0)
    g1[5] = (int)(row_stride_elems & 0xFFFFFFFFu);                          // dim0_stride lo32
    g1[6] = (int)((row_stride_elems >> 32) & 0xFFFFu);                      // dim0_stride hi16
    g1[7] = 0;
    i32x4 z4 = {0, 0, 0, 0};             // groups 2/3: tile_dim3/4 = 0 (unused)
    i32x8 z8 = {0, 0, 0, 0, 0, 0, 0, 0};
    __builtin_amdgcn_tensor_load_to_lds(g0, g1, z4, z4, z8, 0);
}
#endif

// ---------------------------------------------------------------------------
// fp32 [E][K][N] -> bf16 transposed [E][N][K]  (tiled 32x32 transpose)
// Makes the GEMM B operand K-contiguous so WMMA fragments need no LDS shuffle.
// ---------------------------------------------------------------------------
__global__ void transpose_cvt_kernel(const float* __restrict__ W,
                                     __bf16* __restrict__ WT, int K, int Nn) {
    __shared__ float tile[32][33];
    const int e  = blockIdx.z;
    const int k0 = blockIdx.y * 32;
    const int n0 = blockIdx.x * 32;
    const float* Wp  = W  + (size_t)e * K * Nn;
    __bf16*      WTp = WT + (size_t)e * K * Nn;
    const int t = threadIdx.x;     // 256 threads
    {
        int k = t >> 3;            // 0..31
        int n = (t & 7) * 4;       // 0..28
        float4 v = *reinterpret_cast<const float4*>(Wp + (size_t)(k0 + k) * Nn + n0 + n);
        tile[k][n + 0] = v.x; tile[k][n + 1] = v.y;
        tile[k][n + 2] = v.z; tile[k][n + 3] = v.w;
    }
    __syncthreads();
    {
        int n = t >> 3;
        int k = (t & 7) * 4;
        __bf16 o[4];
        o[0] = f2bf(tile[k + 0][n]); o[1] = f2bf(tile[k + 1][n]);
        o[2] = f2bf(tile[k + 2][n]); o[3] = f2bf(tile[k + 3][n]);
        unsigned long long pk;
        __builtin_memcpy(&pk, o, 8);
        *reinterpret_cast<unsigned long long*>(WTp + (size_t)(n0 + n) * K + k0 + k) = pk;
    }
}

// ---------------------------------------------------------------------------
// Router: one wave32 per token. logits = x . Wg (D x E), softmax gate, argmax.
// ---------------------------------------------------------------------------
__global__ void router_kernel(const float* __restrict__ x,
                              const float* __restrict__ Wg,
                              int*   __restrict__ expert_idx,
                              float* __restrict__ gate) {
    int token = blockIdx.x * 8 + (threadIdx.x >> 5);   // 8 waves per block
    int lane  = threadIdx.x & 31;
    if (token >= MOE_B * MOE_N) return;
    const float* xr = x + (size_t)token * MOE_D;

    float acc[MOE_E];
#pragma unroll
    for (int e = 0; e < MOE_E; ++e) acc[e] = 0.0f;

    for (int d = lane; d < MOE_D; d += 32) {
        float xv = xr[d];
        const float* wr = Wg + (size_t)d * MOE_E;
#pragma unroll
        for (int e = 0; e < MOE_E; ++e) acc[e] += xv * wr[e];
    }
#pragma unroll
    for (int e = 0; e < MOE_E; ++e) {
        float v = acc[e];
#pragma unroll
        for (int off = 16; off > 0; off >>= 1) v += __shfl_xor(v, off, 32);
        acc[e] = v;
    }
    if (lane == 0) {
        float m = acc[0]; int bi = 0;
#pragma unroll
        for (int e = 1; e < MOE_E; ++e)
            if (acc[e] > m) { m = acc[e]; bi = e; }   // first-max, matches jnp.argmax
        float s = 0.0f;
#pragma unroll
        for (int e = 0; e < MOE_E; ++e) s += __expf(acc[e] - m);
        expert_idx[token] = bi;
        gate[token]       = 1.0f / s;                 // exp(lmax-m)/sum = 1/sum
    }
}

// ---------------------------------------------------------------------------
// Per-batch sequential cumsum scan (B=8, trivial cost) -> slot assignment
// ---------------------------------------------------------------------------
__global__ void scan_kernel(const int* __restrict__ expert_idx,
                            int* __restrict__ slot_token,
                            int* __restrict__ expert_count) {
    int b = threadIdx.x;
    if (b >= MOE_B) return;
    int cnt[MOE_E];
#pragma unroll
    for (int e = 0; e < MOE_E; ++e) cnt[e] = 0;
    for (int n = 0; n < MOE_N; ++n) {
        int e = expert_idx[b * MOE_N + n];
        int p = cnt[e]++;
        if (p < MOE_CAP) slot_token[((size_t)(b * MOE_E + e)) * MOE_CAP + p] = n;
    }
#pragma unroll
    for (int e = 0; e < MOE_E; ++e)
        expert_count[b * MOE_E + e] = cnt[e] < MOE_CAP ? cnt[e] : MOE_CAP;
}

// ---------------------------------------------------------------------------
// Gather routed tokens into bf16 expert buffers [B*E, CAP, D], zero-fill rest
// ---------------------------------------------------------------------------
__global__ void gather_kernel(const float* __restrict__ x,
                              const int* __restrict__ slot_token,
                              const int* __restrict__ expert_count,
                              __bf16* __restrict__ Xbf) {
    int row = blockIdx.x;                 // [0, B*E*CAP)
    int be  = row / MOE_CAP;
    int c   = row % MOE_CAP;
    int b   = be / MOE_E;
    __bf16* dst = Xbf + (size_t)row * MOE_D;
    if (c < expert_count[be]) {
        int n = slot_token[row];
        const float* src = x + ((size_t)b * MOE_N + n) * MOE_D;
        for (int d = threadIdx.x; d < MOE_D; d += blockDim.x)
            dst[d] = f2bf(src[d]);
    } else {
        __bf16 z = f2bf(0.0f);
        for (int d = threadIdx.x; d < MOE_D; d += blockDim.x)
            dst[d] = z;
    }
}

// ---------------------------------------------------------------------------
// Batched bf16 WMMA GEMM:  C[bz] (MxN) = A[bz] (MxK, row-major)
//                                      x BT[bz % Bmod] (NxK, pre-transposed)
// Block tile 64(M) x 128(N), 8 waves, each wave computes a 32x32 sub-tile
// (2 A frags + 2 B frags -> 4 v_wmma_f32_16x16x32_bf16 per K-step).
// B tile is DMA'd into LDS by the Tensor Data Mover (padded rows of 40 bf16);
// A tile staged with coalesced b128 loads.  M,N,K multiples of 64/128/32.
// ---------------------------------------------------------------------------
template <bool RELU_BF16>
__global__ void gemm_wmma_tdm_kernel(const __bf16* __restrict__ Aall,
                                     const __bf16* __restrict__ BTall,
                                     void* __restrict__ Call,
                                     int Nn, int K,
                                     long Astride, long Bstride, long Cstride,
                                     int Bmod) {
    __shared__ alignas(16) __bf16 ldsA[64 * 40];    // [m][k], padded rows
    __shared__ alignas(16) __bf16 ldsB[128 * 40];   // [n][k], padded rows

    const int bz = blockIdx.z;
    const __bf16* A  = Aall  + (size_t)bz * Astride;
    const __bf16* BT = BTall + (size_t)(bz % Bmod) * Bstride;

    const int rowBase = blockIdx.y * 64;
    const int colBase = blockIdx.x * 128;
    const int tid  = threadIdx.x;
    const int wave = tid >> 5;
    const int lane = tid & 31;
    const int wm   = (wave >> 2) * 32;   // wave row offset: 0 / 32
    const int wn   = (wave & 3) * 32;    // wave col offset: 0..96
    const int r16  = lane & 15;
    const int hi   = lane >> 4;

    v8f acc00 = {}, acc01 = {}, acc10 = {}, acc11 = {};

    // A staging coords: 64 rows x 32 elems, one 16B chunk per thread
    const int arow = tid >> 2;           // 0..63
    const int achk = (tid & 3) * 8;      // 0,8,16,24
    const __bf16* gA = A + (size_t)(rowBase + arow) * K + achk;
    const __bf16* gBtile = BT + (size_t)colBase * K;   // tile row 0; +k0 per step

#if MOE_HAVE_TDM
    const unsigned ldsB_off = (unsigned)(size_t)(&ldsB[0]);
#else
    const int brow = tid >> 1;           // 0..127
    const int bchk = (tid & 1) * 16;     // 0,16
    const __bf16* gB = gBtile + (size_t)brow * K + bchk;
#endif

    for (int k0 = 0; k0 < K; k0 += 32) {
#if MOE_HAVE_TDM
        if (wave == 0 && lane == 0) {
            // DMA B tile: 32 (K, contiguous) x 128 (N rows, stride K elems)
            tdm_load_tile(ldsB_off, gBtile + k0, 32u, 128u, (unsigned long long)K);
        }
#else
        {
            uint4 v0 = *reinterpret_cast<const uint4*>(gB + k0);
            uint4 v1 = *reinterpret_cast<const uint4*>(gB + k0 + 8);
            *reinterpret_cast<uint4*>(&ldsB[brow * 40 + bchk])     = v0;
            *reinterpret_cast<uint4*>(&ldsB[brow * 40 + bchk + 8]) = v1;
        }
#endif
        // stage A tile (coalesced 16B)
        {
            uint4 va = *reinterpret_cast<const uint4*>(gA + k0);
            *reinterpret_cast<uint4*>(&ldsA[arow * 40 + achk]) = va;
        }
        __builtin_prefetch(gA + k0 + 32, 0, 1);
#if MOE_HAVE_TDM
        __builtin_amdgcn_s_wait_tensorcnt(0);   // no-op for waves with count 0
#endif
        __syncthreads();

        // A fragments (16x32): lane<16 -> K 0..15; lane>=16 -> K 16..31
        v16bf a0, a1, b0, b1;
        __builtin_memcpy(&a0, &ldsA[(wm + r16) * 40 + hi * 16], 32);
        __builtin_memcpy(&a1, &ldsA[(wm + 16 + r16) * 40 + hi * 16], 32);
        // B fragments (32x16): lane<16 -> K 0..15 of col; lane>=16 -> K 16..31
        __builtin_memcpy(&b0, &ldsB[(wn + r16) * 40 + hi * 16], 32);
        __builtin_memcpy(&b1, &ldsB[(wn + 16 + r16) * 40 + hi * 16], 32);

        acc00 = __builtin_amdgcn_wmma_f32_16x16x32_bf16(false, a0, false, b0,
                                                        (short)0, acc00, false, false);
        acc01 = __builtin_amdgcn_wmma_f32_16x16x32_bf16(false, a0, false, b1,
                                                        (short)0, acc01, false, false);
        acc10 = __builtin_amdgcn_wmma_f32_16x16x32_bf16(false, a1, false, b0,
                                                        (short)0, acc10, false, false);
        acc11 = __builtin_amdgcn_wmma_f32_16x16x32_bf16(false, a1, false, b1,
                                                        (short)0, acc11, false, false);
        __syncthreads();
    }

    // D layout: VGPR r -> (M = base + r + hi*8, N = base + r16)
    const int row0 = rowBase + wm + hi * 8;
    const int col0 = colBase + wn + r16;
    if (RELU_BF16) {
        __bf16* C = reinterpret_cast<__bf16*>(Call) + (size_t)bz * Cstride;
#pragma unroll
        for (int r = 0; r < 8; ++r) {
            float v00 = acc00[r]; v00 = v00 > 0.0f ? v00 : 0.0f;
            float v01 = acc01[r]; v01 = v01 > 0.0f ? v01 : 0.0f;
            float v10 = acc10[r]; v10 = v10 > 0.0f ? v10 : 0.0f;
            float v11 = acc11[r]; v11 = v11 > 0.0f ? v11 : 0.0f;
            C[(size_t)(row0 + r)      * Nn + col0]      = f2bf(v00);
            C[(size_t)(row0 + r)      * Nn + col0 + 16] = f2bf(v01);
            C[(size_t)(row0 + 16 + r) * Nn + col0]      = f2bf(v10);
            C[(size_t)(row0 + 16 + r) * Nn + col0 + 16] = f2bf(v11);
        }
    } else {
        float* C = reinterpret_cast<float*>(Call) + (size_t)bz * Cstride;
#pragma unroll
        for (int r = 0; r < 8; ++r) {
            C[(size_t)(row0 + r)      * Nn + col0]      = acc00[r];
            C[(size_t)(row0 + r)      * Nn + col0 + 16] = acc01[r];
            C[(size_t)(row0 + 16 + r) * Nn + col0]      = acc10[r];
            C[(size_t)(row0 + 16 + r) * Nn + col0 + 16] = acc11[r];
        }
    }
}

// ---------------------------------------------------------------------------
// Zero the output (unrouted tokens must be zero)
// ---------------------------------------------------------------------------
__global__ void zero_kernel(float4* __restrict__ p, int n4) {
    int i = blockIdx.x * blockDim.x + threadIdx.x;
    if (i < n4) p[i] = float4{0.0f, 0.0f, 0.0f, 0.0f};
}

// ---------------------------------------------------------------------------
// Combine: out[b, n, :] = gate[b,n] * Y[b,e,c,:] for routed slots
// ---------------------------------------------------------------------------
__global__ void combine_kernel(const float* __restrict__ Y,
                               const int* __restrict__ slot_token,
                               const int* __restrict__ expert_count,
                               const float* __restrict__ gate,
                               float* __restrict__ out) {
    int row = blockIdx.x;                 // [0, B*E*CAP)
    int be  = row / MOE_CAP;
    int c   = row % MOE_CAP;
    if (c >= expert_count[be]) return;
    int b = be / MOE_E;
    int n = slot_token[row];
    float g = gate[b * MOE_N + n];
    const float* src = Y + (size_t)row * MOE_D;
    float* dst = out + ((size_t)b * MOE_N + n) * MOE_D;
    int d = threadIdx.x * 4;              // 256 threads * 4 = D
    float4 v = *reinterpret_cast<const float4*>(src + d);
    float4 o; o.x = v.x * g; o.y = v.y * g; o.z = v.z * g; o.w = v.w * g;
    *reinterpret_cast<float4*>(dst + d) = o;
}

// ---------------------------------------------------------------------------
extern "C" void kernel_launch(void* const* d_in, const int* in_sizes, int n_in,
                              void* d_out, int out_size, void* d_ws, size_t ws_size,
                              hipStream_t stream) {
    (void)in_sizes; (void)n_in; (void)out_size; (void)ws_size;
    const float* token_inputs = (const float*)d_in[0];   // [B,N,D]
    const float* Wg           = (const float*)d_in[1];   // [D,E]
    const float* W1           = (const float*)d_in[2];   // [E,D,DFF]
    const float* W2           = (const float*)d_in[3];   // [E,DFF,D]
    float* out = (float*)d_out;                          // [B,N,D]

    // workspace carve-out (256B aligned slices)
    char* ws = (char*)d_ws;
    auto alloc = [&](size_t bytes) -> char* {
        char* p = ws;
        ws += (bytes + 255) & ~(size_t)255;
        return p;
    };
    const size_t nW    = (size_t)MOE_E * MOE_D * MOE_DFF;     // 33.5M each
    const size_t nRows = (size_t)MOE_B * MOE_E * MOE_CAP;     // 20480

    __bf16* W1T = (__bf16*)alloc(nW * sizeof(__bf16));        // [E][DFF][D]
    __bf16* W2T = (__bf16*)alloc(nW * sizeof(__bf16));        // [E][D][DFF]
    __bf16* Xbf = (__bf16*)alloc(nRows * MOE_D   * sizeof(__bf16));
    __bf16* H   = (__bf16*)alloc(nRows * MOE_DFF * sizeof(__bf16));
    float*  Y   = (float*) alloc(nRows * MOE_D   * sizeof(float));
    int*    expert_idx   = (int*)  alloc((size_t)MOE_B * MOE_N * sizeof(int));
    float*  gate         = (float*)alloc((size_t)MOE_B * MOE_N * sizeof(float));
    int*    slot_token   = (int*)  alloc(nRows * sizeof(int));
    int*    expert_count = (int*)  alloc((size_t)MOE_B * MOE_E * sizeof(int));

    // 1) weight conversion + transpose to bf16 (one-time, makes B K-contiguous)
    {
        dim3 g1(MOE_DFF / 32, MOE_D / 32, MOE_E);   // W1 [E][D][DFF] -> W1T [E][DFF][D]
        transpose_cvt_kernel<<<g1, 256, 0, stream>>>(W1, W1T, MOE_D, MOE_DFF);
        dim3 g2(MOE_D / 32, MOE_DFF / 32, MOE_E);   // W2 [E][DFF][D] -> W2T [E][D][DFF]
        transpose_cvt_kernel<<<g2, 256, 0, stream>>>(W2, W2T, MOE_DFF, MOE_D);
    }

    // 2) router (one wave32 per token, 8 waves per block)
    router_kernel<<<(MOE_B * MOE_N) / 8, 256, 0, stream>>>(token_inputs, Wg,
                                                           expert_idx, gate);
    // 3) capacity scan
    scan_kernel<<<1, 32, 0, stream>>>(expert_idx, slot_token, expert_count);

    // 4) gather into bf16 expert buffers
    gather_kernel<<<(int)nRows, 256, 0, stream>>>(token_inputs, slot_token,
                                                  expert_count, Xbf);

    // 5) GEMM1: H = relu(X @ W1)  -> [CAP x DFF] per (b,e)
    {
        dim3 grid(MOE_DFF / 128, MOE_CAP / 64, MOE_B * MOE_E);
        gemm_wmma_tdm_kernel<true><<<grid, 256, 0, stream>>>(
            Xbf, W1T, (void*)H, MOE_DFF, MOE_D,
            (long)MOE_CAP * MOE_D, (long)MOE_D * MOE_DFF,
            (long)MOE_CAP * MOE_DFF, MOE_E);
    }
    // 6) GEMM2: Y = H @ W2  -> [CAP x D] per (b,e)
    {
        dim3 grid(MOE_D / 128, MOE_CAP / 64, MOE_B * MOE_E);
        gemm_wmma_tdm_kernel<false><<<grid, 256, 0, stream>>>(
            H, W2T, (void*)Y, MOE_D, MOE_DFF,
            (long)MOE_CAP * MOE_DFF, (long)MOE_DFF * MOE_D,
            (long)MOE_CAP * MOE_D, MOE_E);
    }

    // 7) zero output, then weighted scatter-combine
    {
        int n4 = (MOE_B * MOE_N * MOE_D) / 4;
        zero_kernel<<<(n4 + 255) / 256, 256, 0, stream>>>((float4*)out, n4);
    }
    combine_kernel<<<(int)nRows, 256, 0, stream>>>(Y, slot_token, expert_count,
                                                   gate, out);
}
